// BerryPhaseRecurrentManifold_56126632624482
// MI455X (gfx1250) — matrix operand order repaired and checked by reading, not compiled, not run on hardware
//
#include <hip/hip_runtime.h>
#include <cstdint>

// ---------------------------------------------------------------------------
// BerryPhaseRecurrentManifold for MI455X (gfx1250, wave32, WMMA + TDM)
//   B=4, S=2048, D=1024, H=1024
//   GEMM1: V = x @ W_in^T + b_in          (M=8192, K=1024, N=3072)  f16 WMMA
//   map  : q = [cos|theta|, sin*axis]     elementwise, f32
//   scan : m <- normalize(q * m)          sequential over S, f32, states -> f16
//   GEMM2: out = states @ W_out^T + b_out (M=8192, K=4096, N=1024)  f16 WMMA
// GEMM tiles are staged into double-buffered LDS by the Tensor Data Mover
// (tensor_load_to_lds, TENSORcnt), overlapping DMA with WMMA compute.
// ---------------------------------------------------------------------------

typedef __attribute__((ext_vector_type(16))) _Float16 v16h;
typedef __attribute__((ext_vector_type(8)))  _Float16 v8h;
typedef __attribute__((ext_vector_type(4)))  _Float16 v4h;
typedef __attribute__((ext_vector_type(8)))  float    v8f;
typedef __attribute__((ext_vector_type(4)))  unsigned int u32x4;
typedef __attribute__((ext_vector_type(8)))  int          i32x8;
typedef __attribute__((ext_vector_type(4)))  int          i32x4;

#define TILE_M 128
#define TILE_N 128
#define TILE_K 32
#define LDSL   40   // halves per LDS row: 32 + 8 pad (16B) -> 80B row stride

#if defined(__has_builtin)
#if __has_builtin(__builtin_amdgcn_tensor_load_to_lds)
#define USE_TDM 1
#endif
#endif
#ifndef USE_TDM
#define USE_TDM 0
#endif

// ---------------------------- f32 -> f16 cast ------------------------------
__global__ __launch_bounds__(256)
void cvt_f32_to_f16_x4(const float4* __restrict__ src, v4h* __restrict__ dst, int n4) {
    int i = blockIdx.x * blockDim.x + threadIdx.x;
    if (i < n4) {
        float4 s = src[i];
        v4h d;
        d[0] = (_Float16)s.x; d[1] = (_Float16)s.y;
        d[2] = (_Float16)s.z; d[3] = (_Float16)s.w;
        dst[i] = d;
    }
}

#if USE_TDM
// DMA one 128-row x 32-half tile (row stride K halves) into LDS at ldsAddr,
// inserting 16B of LDS padding after every 64B row => row stride 80B (LDSL).
__device__ __forceinline__ void tdm_load_tile(unsigned int ldsAddr,
                                              const _Float16* gptr, int K) {
    unsigned long long ga = (unsigned long long)(uintptr_t)gptr;
    u32x4 g0;
    g0[0] = 1u;                                   // count=1 valid descriptor
    g0[1] = ldsAddr;                              // LDS byte address
    g0[2] = (unsigned int)ga;                     // global_addr[31:0]
    g0[3] = (unsigned int)(ga >> 32) | 0x80000000u; // addr[56:32] | type=2

    i32x8 g1;
    g1[0] = (int)(0x10000u        // data_size=1 (2 bytes), workgroup_mask=0
                | (1u << 20)      // pad_enable
                | (3u << 22)      // pad_interval: 16 DWORDs (64B)
                | (3u << 25));    // pad_amount:   4 DWORDs (16B)
    g1[1] = (int)(32u << 16);     // tensor_dim0 = 32 (bits 79:48 low half)
    g1[2] = (int)(128u << 16);    // tensor_dim1 = 128 (bits 111:80 low half)
    g1[3] = (int)(32u << 16);     // tile_dim0 = 32 (bits 127:112)
    g1[4] = 128;                  // tile_dim1 = 128; tile_dim2 = 0
    g1[5] = K;                    // tensor_dim0_stride[31:0] = K elements
    g1[6] = 0;                    // stride[47:32]=0; tensor_dim1_stride lo=0
    g1[7] = 0;                    // tensor_dim1_stride hi = 0
    i32x4 z4 = {0, 0, 0, 0};      // groups 2/3: dims 3..4 unused
    i32x8 z8 = {0, 0, 0, 0, 0, 0, 0, 0};
    // clang-23 / therock-10.0 6-arg form
    __builtin_amdgcn_tensor_load_to_lds(g0, g1, z4, z4, z8, 0);
}
#endif

// ------------------------- NT GEMM via WMMA f16 ----------------------------
// C[m,n] = sum_k A[m,k]*Bm[n,k] + bias[n]
// A: (M,K) f16 row-major; Bm: (N,K) f16 row-major; C: (M,N) f32.
// 256 threads = 8 waves (4 along M x 2 along N); each wave 32x64 = 2x4 frags.
__global__ __launch_bounds__(256)
void gemm_f16f32_nt(const _Float16* __restrict__ A, const _Float16* __restrict__ Bm,
                    const float* __restrict__ bias, float* __restrict__ C,
                    int M, int N, int K) {
    __shared__ _Float16 lA[2][TILE_M * LDSL];
    __shared__ _Float16 lB[2][TILE_N * LDSL];

    const int tid    = threadIdx.x;
    const int bm     = blockIdx.y * TILE_M;
    const int bn     = blockIdx.x * TILE_N;
    const int wave   = tid >> 5;
    const int lane   = tid & 31;
    const int laneLo = lane & 15;    // A: M row | B/D: N column
    const int laneHi = lane >> 4;    // selects K-half / M-half
    const int wm     = (wave & 3) * 32;   // wave M offset in tile
    const int wn     = (wave >> 2) * 64;  // wave N offset in tile

    v8f acc[2][4];
#pragma unroll
    for (int mi = 0; mi < 2; ++mi)
#pragma unroll
        for (int ni = 0; ni < 4; ++ni)
            acc[mi][ni] = (v8f)0.0f;

    // fragment loads + 8 WMMA on one staged K-tile
    auto mma_tile = [&](const _Float16* tA, const _Float16* tB) {
        // A frag (16x32): lane<16 -> K 0..7 & 16..23; lane>=16 -> K 8..15 & 24..31
        v16h af[2];
#pragma unroll
        for (int mi = 0; mi < 2; ++mi) {
            const _Float16* p = &tA[(wm + mi * 16 + laneLo) * LDSL];
            v8h lo = *(const v8h*)(p + laneHi * 8);
            v8h hi = *(const v8h*)(p + 16 + laneHi * 8);
            af[mi] = __builtin_shufflevector(lo, hi, 0,1,2,3,4,5,6,7,8,9,10,11,12,13,14,15);
        }
        // B frag (32x16): lane<16 -> N=lane,K 0..7 & 16..23; lane>=16 -> N=lane-16,K 8..15 & 24..31
        v16h bf[4];
#pragma unroll
        for (int ni = 0; ni < 4; ++ni) {
            const _Float16* p = &tB[(wn + ni * 16 + laneLo) * LDSL];
            v8h lo = *(const v8h*)(p + laneHi * 8);
            v8h hi = *(const v8h*)(p + 16 + laneHi * 8);
            bf[ni] = __builtin_shufflevector(lo, hi, 0,1,2,3,4,5,6,7,8,9,10,11,12,13,14,15);
        }
#pragma unroll
        for (int mi = 0; mi < 2; ++mi)
#pragma unroll
            for (int ni = 0; ni < 4; ++ni)
                acc[mi][ni] = __builtin_amdgcn_wmma_f32_16x16x32_f16(
                    false, af[mi], false, bf[ni], (short)0, acc[mi][ni], false, false);
    };

#if USE_TDM
    const int KT = K / TILE_K;
    // prologue: DMA tile 0 into buffer 0 (wave0 -> A, wave1 -> B)
    if (wave == 0) {
        tdm_load_tile((unsigned int)(uintptr_t)&lA[0][0], A + (size_t)bm * K, K);
        __builtin_amdgcn_s_wait_tensorcnt(0);
    }
    if (wave == 1) {
        tdm_load_tile((unsigned int)(uintptr_t)&lB[0][0], Bm + (size_t)bn * K, K);
        __builtin_amdgcn_s_wait_tensorcnt(0);
    }
    __syncthreads();

    for (int kt = 0; kt < KT; ++kt) {
        const int cur = kt & 1;
        const int nxt = cur ^ 1;
        if (kt + 1 < KT) {   // prefetch next tile via TDM while computing
            if (wave == 0)
                tdm_load_tile((unsigned int)(uintptr_t)&lA[nxt][0],
                              A + (size_t)bm * K + (size_t)(kt + 1) * TILE_K, K);
            if (wave == 1)
                tdm_load_tile((unsigned int)(uintptr_t)&lB[nxt][0],
                              Bm + (size_t)bn * K + (size_t)(kt + 1) * TILE_K, K);
        }
        mma_tile(&lA[cur][0], &lB[cur][0]);
        if (wave < 2) __builtin_amdgcn_s_wait_tensorcnt(0);
        __syncthreads();   // publish next buffer / retire reads of it
    }
#else
    // fallback: synchronous cooperative staging (single buffer)
    for (int k0 = 0; k0 < K; k0 += TILE_K) {
#pragma unroll
        for (int i = 0; i < 2; ++i) {
            int id  = tid + i * 256;         // 0..511
            int row = id >> 2;               // 0..127
            int c8  = (id & 3) * 8;          // half-offset within K-tile
            *(uint4*)(&lA[0][row * LDSL + c8]) =
                *(const uint4*)(&A[(size_t)(bm + row) * K + k0 + c8]);
            *(uint4*)(&lB[0][row * LDSL + c8]) =
                *(const uint4*)(&Bm[(size_t)(bn + row) * K + k0 + c8]);
        }
        __syncthreads();
        mma_tile(&lA[0][0], &lB[0][0]);
        __syncthreads();
    }
#endif

    // D layout: VGPR r -> row = laneHi*8 + r, col = laneLo
#pragma unroll
    for (int mi = 0; mi < 2; ++mi) {
        int rowBase = bm + wm + mi * 16 + laneHi * 8;
#pragma unroll
        for (int ni = 0; ni < 4; ++ni) {
            int col  = bn + wn + ni * 16 + laneLo;
            float bv = bias[col];
#pragma unroll
            for (int r = 0; r < 8; ++r)
                C[(size_t)(rowBase + r) * N + col] = acc[mi][ni][r] + bv;
        }
    }
}

// ------------------- SU(2) exponential map: v -> quaternion ----------------
__global__ __launch_bounds__(256)
void quatmap_kernel(const float* __restrict__ V, float* __restrict__ Q,
                    int total, int Hc) {
    int i = blockIdx.x * blockDim.x + threadIdx.x;   // over B*S*H
    if (i >= total) return;
    int bs = i / Hc;
    int h  = i - bs * Hc;
    const float* v = &V[(size_t)bs * 3 * Hc + 3 * h];
    float vx = v[0], vy = v[1], vz = v[2];
    float th = sqrtf(vx * vx + vy * vy + vz * vz);
    float sc = sinf(th) / (th + 1e-8f);             // sin(theta)*axis = sc*v
    float4 q = make_float4(cosf(th), vx * sc, vy * sc, vz * sc);
    *(float4*)&Q[(size_t)i * 4] = q;
}

// ------------------- sequential quaternion recurrence ----------------------
__global__ __launch_bounds__(256)
void scan_kernel(const float* __restrict__ Q, _Float16* __restrict__ Sh,
                 float* __restrict__ mfinal, int Bc, int Sc, int Hc) {
    int idx = blockIdx.x * blockDim.x + threadIdx.x;   // 0..B*H-1
    if (idx >= Bc * Hc) return;
    int b = idx / Hc;
    int h = idx - b * Hc;

    float mw = 1.0f, mx = 0.0f, my = 0.0f, mz = 0.0f;
    for (int s = 0; s < Sc; ++s) {
        const float4 q = *(const float4*)&Q[(((size_t)b * Sc + s) * Hc + h) * 4];
        float w1 = q.x, x1 = q.y, y1 = q.z, z1 = q.w;
        float w = w1 * mw - x1 * mx - y1 * my - z1 * mz;
        float x = w1 * mx + x1 * mw + y1 * mz - z1 * my;
        float y = w1 * my - x1 * mz + y1 * mw + z1 * mx;
        float z = w1 * mz + x1 * my - y1 * mx + z1 * mw;
        float n = sqrtf(w * w + x * x + y * y + z * z);
        float r = 1.0f / (n + 1e-8f);
        mw = w * r; mx = x * r; my = y * r; mz = z * r;
        v4h st;
        st[0] = (_Float16)mw; st[1] = (_Float16)mx;
        st[2] = (_Float16)my; st[3] = (_Float16)mz;
        *(v4h*)&Sh[((size_t)b * Sc + s) * 4 * Hc + 4 * h] = st;
    }
    float4 mf = make_float4(mw, mx, my, mz);
    *(float4*)&mfinal[(size_t)idx * 4] = mf;   // (B,H,4) flat
}

// ---------------------------------------------------------------------------
extern "C" void kernel_launch(void* const* d_in, const int* in_sizes, int n_in,
                              void* d_out, int out_size, void* d_ws, size_t ws_size,
                              hipStream_t stream) {
    const float* x     = (const float*)d_in[0];   // (4,2048,1024)
    const float* W_in  = (const float*)d_in[1];   // (3072,1024)
    const float* b_in  = (const float*)d_in[2];   // (3072,)
    const float* W_out = (const float*)d_in[3];   // (1024,4096)
    const float* b_out = (const float*)d_in[4];   // (1024,)
    float* out = (float*)d_out;                   // [full_output | m_final]

    const int Bc = 4, Sc = 2048, Dc = 1024, Hc = 1024;
    const int M  = Bc * Sc;        // 8192
    const int N1 = 3 * Hc;         // 3072
    const int K2 = 4 * Hc;         // 4096

    // workspace carve-up (bytes)
    char* ws = (char*)d_ws;
    _Float16* Xh  = (_Float16*)(ws);                                   // 16 MB
    _Float16* W1h = (_Float16*)(ws + (size_t)16 * 1024 * 1024);        //  6 MB
    _Float16* W2h = (_Float16*)(ws + (size_t)22 * 1024 * 1024);        //  8 MB
    float*    V   = (float*)   (ws + (size_t)30 * 1024 * 1024);        // 96 MB
    float*    Q   = (float*)   (ws + (size_t)126 * 1024 * 1024);       // 128 MB
    _Float16* Sh  = (_Float16*)(ws + (size_t)254 * 1024 * 1024);       // 64 MB -> 318 MB

    // 1) casts to f16
    {
        int n4;
        n4 = (M * Dc) / 4;
        cvt_f32_to_f16_x4<<<(n4 + 255) / 256, 256, 0, stream>>>((const float4*)x, (v4h*)Xh, n4);
        n4 = (N1 * Dc) / 4;
        cvt_f32_to_f16_x4<<<(n4 + 255) / 256, 256, 0, stream>>>((const float4*)W_in, (v4h*)W1h, n4);
        n4 = (Dc * K2) / 4;
        cvt_f32_to_f16_x4<<<(n4 + 255) / 256, 256, 0, stream>>>((const float4*)W_out, (v4h*)W2h, n4);
    }

    // 2) GEMM1: V = Xh * W1h^T + b_in   (M x N1, K = Dc)
    {
        dim3 grid(N1 / TILE_N, M / TILE_M);   // (24, 64)
        gemm_f16f32_nt<<<grid, 256, 0, stream>>>(Xh, W1h, b_in, V, M, N1, Dc);
    }

    // 3) quaternion exponential map
    {
        int total = M * Hc;
        quatmap_kernel<<<(total + 255) / 256, 256, 0, stream>>>(V, Q, total, Hc);
    }

    // 4) sequential scan; m_final goes to tail of d_out
    {
        float* mfinal = out + (size_t)M * Dc;
        int nch = Bc * Hc;
        scan_kernel<<<(nch + 255) / 256, 256, 0, stream>>>(Q, Sh, mfinal, Bc, Sc, Hc);
    }

    // 5) GEMM2: out = Sh * W2h^T + b_out  (M x Dc, K = 4H)
    {
        dim3 grid(Dc / TILE_N, M / TILE_M);   // (8, 64)
        gemm_f16f32_nt<<<grid, 256, 0, stream>>>(Sh, W2h, b_out, out, M, Dc, K2);
    }
}
